// SASRecBlock_66245575573896
// MI455X (gfx1250) — matrix operand order repaired and checked by reading, not compiled, not run on hardware
//
#include <hip/hip_runtime.h>
#include <hip/hip_bf16.h>
#include <math.h>

typedef __attribute__((ext_vector_type(16))) _Float16 v16h;
typedef __attribute__((ext_vector_type(8)))  float    v8f;
typedef __attribute__((ext_vector_type(4)))  unsigned int u32x4;
typedef __attribute__((ext_vector_type(8)))  unsigned int u32x8;

#define USE_TDM 1

#define B_   32
#define S_   512
#define H_   512
#define NH_  8
#define HD_  64
#define FF_  2048
#define MTOK (B_ * S_)   // 16384 tokens

// ---------------- WMMA fragment helpers (ISA 7.12.2 layouts, wave32) ----------------
// All LDS layouts arranged so each lane's fragment data is contiguous 16B-aligned
// chunks -> ds_load_b128, no repacking.

__device__ inline v16h lds_load_a_frag(const _Float16* base, int ld) {
  const int lane = threadIdx.x & 31;
  const _Float16* p = base + (lane & 15) * ld + ((lane >> 4) << 3);
  v16h a;
#pragma unroll
  for (int j = 0; j < 8; ++j) a[j]     = p[j];
#pragma unroll
  for (int j = 0; j < 8; ++j) a[8 + j] = p[16 + j];
  return a;
}

// B fragment from a TRANSPOSED tile baseT[n][k]; two 8-half chunks per lane (k = 8h.., 16+8h..).
__device__ inline v16h lds_load_bT_frag(const _Float16* baseT, int ld) {
  const int lane = threadIdx.x & 31;
  const _Float16* p = baseT + (lane & 15) * ld + ((lane >> 4) << 3);
  v16h b;
#pragma unroll
  for (int j = 0; j < 8; ++j) b[j]     = p[j];
#pragma unroll
  for (int j = 0; j < 8; ++j) b[8 + j] = p[16 + j];
  return b;
}

__device__ inline float rowmax16(float v) {
#pragma unroll
  for (int m = 8; m >= 1; m >>= 1) v = fmaxf(v, __shfl_xor(v, m, 32));
  return v;
}
__device__ inline float rowsum16(float v) {
#pragma unroll
  for (int m = 8; m >= 1; m >>= 1) v += __shfl_xor(v, m, 32);
  return v;
}

// ---------------- Tiled WMMA GEMM:  C[M,N] = A[M,K] @ W[K,N] + bias ----------------

#define BM 128
#define BN 64
#define BK 32
#define LDA  (BK + 8)
#define LDBT (BK + 8)

#define EPI_NONE 0
#define EPI_GELU 1
#define EPI_QKV  2

__global__ __launch_bounds__(256) void gemm_wmma_kernel(
    const float* __restrict__ A, const float* __restrict__ W,
    const float* __restrict__ bias,
    float* __restrict__ Cf, _Float16* __restrict__ Cqkv,
    int M, int K, int N, int epi)
{
  __shared__ _Float16 sA [2][BM * LDA];
  __shared__ _Float16 sBt[2][BN * LDBT];

  const int tid  = threadIdx.x;
  const int lane = tid & 31;
  const int wave = tid >> 5;
  const int wm   = (wave >> 1) * 32;
  const int wn   = (wave & 1)  * 32;
  const int m0   = blockIdx.x * BM;
  const int n0   = blockIdx.y * BN;

  // Load all staging vectors into registers first (all global loads in flight),
  // then convert+store -> LOADcnt drains incrementally instead of wait-0 per load.
  auto load_tile = [&](int kt, int buf) {
    float4 av[4], wv[2];
#pragma unroll
    for (int i = 0; i < 4; ++i) {
      int e = tid + i * 256, r = e >> 3, c4 = (e & 7) * 4;
      av[i] = *(const float4*)&A[(size_t)(m0 + r) * K + kt * BK + c4];
    }
#pragma unroll
    for (int i = 0; i < 2; ++i) {
      int e = tid + i * 256, r = e >> 4, c4 = (e & 15) * 4;
      wv[i] = *(const float4*)&W[(size_t)(kt * BK + r) * N + n0 + c4];
    }
#pragma unroll
    for (int i = 0; i < 4; ++i) {
      int e = tid + i * 256, r = e >> 3, c4 = (e & 7) * 4;
      _Float16* dst = &sA[buf][r * LDA + c4];
      dst[0] = (_Float16)av[i].x; dst[1] = (_Float16)av[i].y;
      dst[2] = (_Float16)av[i].z; dst[3] = (_Float16)av[i].w;
    }
#pragma unroll
    for (int i = 0; i < 2; ++i) {
      int e = tid + i * 256, r = e >> 4, c4 = (e & 15) * 4;
      sBt[buf][(c4 + 0) * LDBT + r] = (_Float16)wv[i].x;
      sBt[buf][(c4 + 1) * LDBT + r] = (_Float16)wv[i].y;
      sBt[buf][(c4 + 2) * LDBT + r] = (_Float16)wv[i].z;
      sBt[buf][(c4 + 3) * LDBT + r] = (_Float16)wv[i].w;
    }
  };

  v8f zero = {};
  v8f acc[2][2];
#pragma unroll
  for (int i = 0; i < 2; ++i)
#pragma unroll
    for (int j = 0; j < 2; ++j) acc[i][j] = zero;

  const int kIters = K / BK;
  load_tile(0, 0);
  __syncthreads();

  for (int kt = 0; kt < kIters; ++kt) {
    const int buf = kt & 1;
    if (kt + 2 < kIters) {   // L2 prefetch two tiles ahead (global_prefetch_b8)
      __builtin_prefetch(&A[(size_t)(m0 + (tid >> 1)) * K + (kt + 2) * BK + (tid & 1) * 16], 0, 3);
      __builtin_prefetch(&W[(size_t)((kt + 2) * BK + (tid >> 3)) * N + n0 + (tid & 7) * 8], 0, 3);
    }
    if (kt + 1 < kIters) load_tile(kt + 1, buf ^ 1);  // overlap with WMMAs below

    v16h a0 = lds_load_a_frag (sA [buf] + (wm + 0)  * LDA, LDA);
    v16h a1 = lds_load_a_frag (sA [buf] + (wm + 16) * LDA, LDA);
    v16h b0 = lds_load_bT_frag(sBt[buf] + (wn + 0)  * LDBT, LDBT);
    v16h b1 = lds_load_bT_frag(sBt[buf] + (wn + 16) * LDBT, LDBT);

    acc[0][0] = __builtin_amdgcn_wmma_f32_16x16x32_f16(false, a0, false, b0, (short)0, acc[0][0], false, false);
    acc[0][1] = __builtin_amdgcn_wmma_f32_16x16x32_f16(false, a0, false, b1, (short)0, acc[0][1], false, false);
    acc[1][0] = __builtin_amdgcn_wmma_f32_16x16x32_f16(false, a1, false, b0, (short)0, acc[1][0], false, false);
    acc[1][1] = __builtin_amdgcn_wmma_f32_16x16x32_f16(false, a1, false, b1, (short)0, acc[1][1], false, false);

    __syncthreads();
  }

  const int halfrow = (lane & 16) >> 1;
  const int cn      = lane & 15;
#pragma unroll
  for (int mi = 0; mi < 2; ++mi)
#pragma unroll
    for (int ni = 0; ni < 2; ++ni)
#pragma unroll
      for (int r = 0; r < 8; ++r) {
        int row = m0 + wm + mi * 16 + halfrow + r;
        int col = n0 + wn + ni * 16 + cn;
        float v = acc[mi][ni][r] + bias[col];
        if (epi == EPI_GELU) v = 0.5f * v * (1.0f + erff(v * 0.70710678f));
        if (epi == EPI_QKV) {
          int bb = row / S_, ss = row % S_;
          int hh = col >> 6, dd = col & 63;
          Cqkv[(((size_t)(bb * NH_ + hh) * S_) + ss) * HD_ + dd] = (_Float16)v;
        } else {
          Cf[(size_t)row * N + col] = v;
        }
      }
}

// ---------------- Flash attention: one wave per 16-query tile, 32-key chunks ----------------

#define AK_LD  72   // 144 B row pitch; matches TDM pad: 32 dwords data + 4 dwords pad
#define AVT_LD 40
#define AP_LD  40

__global__ __launch_bounds__(256) void attn_wmma_kernel(
    const _Float16* __restrict__ Qh, const _Float16* __restrict__ Kh,
    const _Float16* __restrict__ Vh, const int* __restrict__ mask,
    float* __restrict__ ctx)
{
  __shared__ _Float16 sK [32 * AK_LD];
  __shared__ _Float16 sVt[64 * AVT_LD];
  __shared__ _Float16 sP [8][16 * AP_LD];
  __shared__ int smask[32];

  const int tid  = threadIdx.x;
  const int lane = tid & 31;
  const int wave = tid >> 5;

  const int qblk = blockIdx.x & 3;
  const int h    = (blockIdx.x >> 2) & (NH_ - 1);
  const int b    = blockIdx.x >> 5;
  const int q0   = qblk * 128 + wave * 16;

  const _Float16* qrow = Qh + (((size_t)(b * NH_ + h) * S_) + q0 + (lane & 15)) * HD_;
  const int kh = (lane >> 4) << 3;
  v16h qa[2];
#pragma unroll
  for (int f = 0; f < 2; ++f) {
    union { uint4 u; _Float16 hv[8]; } t0, t1;
    t0.u = *(const uint4*)(qrow + f * 32 + kh);
    t1.u = *(const uint4*)(qrow + f * 32 + 16 + kh);
#pragma unroll
    for (int j = 0; j < 8; ++j) { qa[f][j] = t0.hv[j]; qa[f][8 + j] = t1.hv[j]; }
  }

  v8f zero = {};
  v8f o[4];
  float mrow[8], lrow[8];
#pragma unroll
  for (int j = 0; j < 4; ++j) o[j] = zero;
#pragma unroll
  for (int r = 0; r < 8; ++r) { mrow[r] = -1e30f; lrow[r] = 0.0f; }

  const _Float16* kbase = Kh + ((size_t)(b * NH_ + h) * S_) * HD_;
  const _Float16* vbase = Vh + ((size_t)(b * NH_ + h) * S_) * HD_;

  for (int c = 0; c < S_ / 32; ++c) {
    __syncthreads();
#if USE_TDM
    // ---- K chunk via Tensor Data Mover: 32x64 f16 2D tile, global -> LDS with
    // row padding 64->72 halfs (pad_interval=32 dwords, pad_amount=4 dwords).
    // Issued once by wave 0 (EXEC ignored by TENSOR ops); others sync at barrier.
    if (wave == 0) {
      unsigned long long ga = (unsigned long long)(uintptr_t)(kbase + (size_t)(c * 32) * HD_);
      unsigned lds_off = (unsigned)(uintptr_t)&sK[0];
      u32x4 g0;
      g0[0] = 1u;                                   // count=1 valid user descriptor
      g0[1] = lds_off;                              // lds_addr (bytes)
      g0[2] = (unsigned)(ga & 0xffffffffu);         // global_addr[31:0]
      g0[3] = (unsigned)((ga >> 32) & 0x01ffffffu)  // global_addr[56:32]
            | (2u << 30);                           // type=2 (image)
      u32x8 g1;
      g1[0] = (1u << 16)        // data_size = 2 bytes
            | (1u << 20)        // pad_enable
            | (4u << 22)        // pad_interval: 32 DWORDs between pads
            | (3u << 25);       // pad_amount: 4 DWORDs
      g1[1] = (64u << 16);      // tensor_dim0 = 64 (bits 79:48, low half)
      g1[2] = (32u << 16);      // tensor_dim1 = 32 (bits 111:80, low half)
      g1[3] = (64u << 16);      // tile_dim0 = 64 (bits 127:112)
      g1[4] = 32u;              // tile_dim1 = 32, tile_dim2 = 0
      g1[5] = 64u;              // tensor_dim0_stride = 64 elements
      g1[6] = 0u;
      g1[7] = 0u;
      asm volatile("tensor_load_to_lds %0, %1" :: "s"(g0), "s"(g1) : "memory");
      __builtin_amdgcn_s_wait_tensorcnt(0);
    }
    {   // V chunk staged manually (TDM cannot transpose): sVt[dim][key]
      const int r  = tid >> 3;
      const int c8 = (tid & 7) * 8;
      union { uint4 u; _Float16 hv[8]; } vv;
      vv.u = *(const uint4*)&vbase[(size_t)(c * 32 + r) * HD_ + c8];
#pragma unroll
      for (int j = 0; j < 8; ++j) sVt[(c8 + j) * AVT_LD + r] = vv.hv[j];
    }
#else
    {
      const int r  = tid >> 3;
      const int c8 = (tid & 7) * 8;
      uint4 kv = *(const uint4*)&kbase[(size_t)(c * 32 + r) * HD_ + c8];
      *(uint4*)&sK[r * AK_LD + c8] = kv;
      union { uint4 u; _Float16 hv[8]; } vv;
      vv.u = *(const uint4*)&vbase[(size_t)(c * 32 + r) * HD_ + c8];
#pragma unroll
      for (int j = 0; j < 8; ++j) sVt[(c8 + j) * AVT_LD + r] = vv.hv[j];
    }
#endif
    if (tid < 32) smask[tid] = mask[b * S_ + c * 32 + tid];
    __syncthreads();

    // ---- scores: S(16x32) = Q(16x64) @ K^T(64x32) ----
    float p[2][8], mnew[8];
#pragma unroll
    for (int t = 0; t < 2; ++t) {
      v16h bK0 = lds_load_bT_frag(sK + (t * 16) * AK_LD,      AK_LD);
      v16h bK1 = lds_load_bT_frag(sK + (t * 16) * AK_LD + 32, AK_LD);
      v8f s = zero;
      s = __builtin_amdgcn_wmma_f32_16x16x32_f16(false, qa[0], false, bK0, (short)0, s, false, false);
      s = __builtin_amdgcn_wmma_f32_16x16x32_f16(false, qa[1], false, bK1, (short)0, s, false, false);
      const int mv = smask[t * 16 + (lane & 15)];
#pragma unroll
      for (int r = 0; r < 8; ++r) {
        float sv = s[r] * 0.125f;
        if (mv == 0) sv = -1e9f;
        p[t][r] = sv;
      }
    }

    // ---- online softmax ----
#pragma unroll
    for (int r = 0; r < 8; ++r)
      mnew[r] = fmaxf(mrow[r], fmaxf(rowmax16(p[0][r]), rowmax16(p[1][r])));
    float scale[8];
#pragma unroll
    for (int r = 0; r < 8; ++r) scale[r] = expf(mrow[r] - mnew[r]);
#pragma unroll
    for (int t = 0; t < 2; ++t)
#pragma unroll
      for (int r = 0; r < 8; ++r) p[t][r] = expf(p[t][r] - mnew[r]);
#pragma unroll
    for (int r = 0; r < 8; ++r) {
      lrow[r] = lrow[r] * scale[r] + rowsum16(p[0][r]) + rowsum16(p[1][r]);
      mrow[r] = mnew[r];
    }
#pragma unroll
    for (int j = 0; j < 4; ++j)
#pragma unroll
      for (int r = 0; r < 8; ++r) o[j][r] *= scale[r];

    // ---- P -> LDS -> A-fragment ----
    {
      const int rowoff = (lane & 16) >> 1;
#pragma unroll
      for (int t = 0; t < 2; ++t)
#pragma unroll
        for (int r = 0; r < 8; ++r)
          sP[wave][(rowoff + r) * AP_LD + t * 16 + (lane & 15)] = (_Float16)p[t][r];
    }
    __syncthreads();
    v16h pa = lds_load_a_frag(sP[wave], AP_LD);

    // ---- ctx += P(16x32) @ V(32x64) ----
#pragma unroll
    for (int j = 0; j < 4; ++j) {
      v16h bV = lds_load_bT_frag(sVt + (j * 16) * AVT_LD, AVT_LD);
      o[j] = __builtin_amdgcn_wmma_f32_16x16x32_f16(false, pa, false, bV, (short)0, o[j], false, false);
    }
  }

  const int rowoff = (lane & 16) >> 1;
  const int n      = lane & 15;
  float linv[8];
#pragma unroll
  for (int r = 0; r < 8; ++r) linv[r] = 1.0f / lrow[r];
#pragma unroll
  for (int j = 0; j < 4; ++j)
#pragma unroll
    for (int r = 0; r < 8; ++r) {
      int qi = q0 + rowoff + r;
      ctx[((size_t)(b * S_) + qi) * H_ + h * HD_ + j * 16 + n] = o[j][r] * linv[r];
    }
}

// ---------------- fused residual-add + LayerNorm ----------------

__global__ __launch_bounds__(256) void add_ln_kernel(
    const float* __restrict__ y, const float* __restrict__ res,
    const float* __restrict__ g, const float* __restrict__ be,
    float* __restrict__ out)
{
  __shared__ float ssum[8], ssq[8];
  const int row = blockIdx.x;
  const int tid = threadIdx.x;
  const float* yr = y   + (size_t)row * H_;
  const float* rr = res + (size_t)row * H_;
  float v0 = yr[tid]       + rr[tid];
  float v1 = yr[tid + 256] + rr[tid + 256];
  float s = v0 + v1, q = v0 * v0 + v1 * v1;
#pragma unroll
  for (int m = 16; m >= 1; m >>= 1) { s += __shfl_xor(s, m, 32); q += __shfl_xor(q, m, 32); }
  if ((tid & 31) == 0) { ssum[tid >> 5] = s; ssq[tid >> 5] = q; }
  __syncthreads();
  float ts = 0.f, tq = 0.f;
#pragma unroll
  for (int i = 0; i < 8; ++i) { ts += ssum[i]; tq += ssq[i]; }
  const float mean = ts * (1.0f / H_);
  const float var  = tq * (1.0f / H_) - mean * mean;
  const float inv  = rsqrtf(var + 1e-5f);
  float* orow = out + (size_t)row * H_;
  orow[tid]       = (v0 - mean) * inv * g[tid]       + be[tid];
  orow[tid + 256] = (v1 - mean) * inv * g[tid + 256] + be[tid + 256];
}

// ---------------- host orchestration ----------------

extern "C" void kernel_launch(void* const* d_in, const int* in_sizes, int n_in,
                              void* d_out, int out_size, void* d_ws, size_t ws_size,
                              hipStream_t stream)
{
  (void)in_sizes; (void)n_in; (void)out_size; (void)ws_size;
  const float* x    = (const float*)d_in[0];
  const int*   mask = (const int*)  d_in[1];
  const float* wq   = (const float*)d_in[2];  const float* bq   = (const float*)d_in[3];
  const float* wk   = (const float*)d_in[4];  const float* bk   = (const float*)d_in[5];
  const float* wv   = (const float*)d_in[6];  const float* bv   = (const float*)d_in[7];
  const float* wo   = (const float*)d_in[8];  const float* bo   = (const float*)d_in[9];
  const float* g1   = (const float*)d_in[10]; const float* b1ln = (const float*)d_in[11];
  const float* w1   = (const float*)d_in[12]; const float* bb1  = (const float*)d_in[13];
  const float* w2   = (const float*)d_in[14]; const float* bb2  = (const float*)d_in[15];
  const float* g2   = (const float*)d_in[16]; const float* b2ln = (const float*)d_in[17];
  float* out = (float*)d_out;

  char* ws = (char*)d_ws;
  size_t off = 0;
  auto carve = [&](size_t bytes) -> char* {
    char* p = ws + off; off += (bytes + 255) & ~(size_t)255; return p;
  };

  char*     qkv_region = carve((size_t)3 * MTOK * H_ * sizeof(_Float16));
  _Float16* Qh = (_Float16*)qkv_region;
  _Float16* Kh = Qh + (size_t)MTOK * H_;
  _Float16* Vh = Kh + (size_t)MTOK * H_;
  float* ctx  = (float*)carve((size_t)MTOK * H_  * sizeof(float));
  float* attn = (float*)carve((size_t)MTOK * H_  * sizeof(float));
  float* hbuf = (float*)carve((size_t)MTOK * FF_ * sizeof(float));
  float* t0 = (float*)qkv_region;
  float* t1 = (float*)qkv_region;

  const dim3 blk(256);
  const dim3 gH(MTOK / BM, H_ / BN);

  gemm_wmma_kernel<<<gH, blk, 0, stream>>>(x, wq, bq, nullptr, Qh, MTOK, H_, H_, EPI_QKV);
  gemm_wmma_kernel<<<gH, blk, 0, stream>>>(x, wk, bk, nullptr, Kh, MTOK, H_, H_, EPI_QKV);
  gemm_wmma_kernel<<<gH, blk, 0, stream>>>(x, wv, bv, nullptr, Vh, MTOK, H_, H_, EPI_QKV);

  attn_wmma_kernel<<<dim3(B_ * NH_ * (S_ / 128)), blk, 0, stream>>>(Qh, Kh, Vh, mask, ctx);

  gemm_wmma_kernel<<<gH, blk, 0, stream>>>(ctx, wo, bo, t0, nullptr, MTOK, H_, H_, EPI_NONE);
  add_ln_kernel<<<dim3(MTOK), blk, 0, stream>>>(t0, x, g1, b1ln, attn);

  gemm_wmma_kernel<<<dim3(MTOK / BM, FF_ / BN), blk, 0, stream>>>(attn, w1, bb1, hbuf, nullptr, MTOK, H_, FF_, EPI_GELU);
  gemm_wmma_kernel<<<gH, blk, 0, stream>>>(hbuf, w2, bb2, t1, nullptr, MTOK, FF_, H_, EPI_NONE);
  add_ln_kernel<<<dim3(MTOK), blk, 0, stream>>>(t1, attn, g2, b2ln, out);
}
